// TransformerBlock_72825465471387
// MI455X (gfx1250) — matrix operand order repaired and checked by reading, not compiled, not run on hardware
//
#include <hip/hip_runtime.h>
#include <hip/hip_bf16.h>
#include <math.h>

// ---------------------------------------------------------------------------
// TransformerBlock for MI455X (gfx1250): f16 WMMA + TDM tensor_load_to_lds
// double-buffered GEMMs.  D_MODEL=1024, H=16, dh=64, D_FF=4096, B=2, T=2048.
// ---------------------------------------------------------------------------

typedef __attribute__((ext_vector_type(16))) _Float16     v16h;
typedef __attribute__((ext_vector_type(8)))  float        v8f;
typedef __attribute__((ext_vector_type(4)))  unsigned int v4u;
typedef __attribute__((ext_vector_type(8)))  int          v8i;
typedef __attribute__((ext_vector_type(4)))  int          v4i;

#define D_MODEL   1024
#define NUM_HEADS 16
#define D_HEAD    64
#define D_FF      4096
#define BB        2
#define TT        2048
#define M_TOK     (BB * TT)       // 4096 token rows
#define EPS       1e-5f

// ------------------------------- utilities ---------------------------------

__global__ void f32_to_f16_kernel(const float* __restrict__ in,
                                  _Float16* __restrict__ out, int n) {
  int i = blockIdx.x * 256 + threadIdx.x;
  if (i < n) out[i] = (_Float16)in[i];
}

// One block per token row: h[row,:] = x/rms(x) * g, emitted as f16.
__global__ __launch_bounds__(256)
void rmsnorm_f16_kernel(const float* __restrict__ x, const float* __restrict__ g,
                        _Float16* __restrict__ out) {
  const int row = blockIdx.x;
  const float* xr = x + (size_t)row * D_MODEL;
  float s = 0.f;
  for (int c = threadIdx.x; c < D_MODEL; c += 256) { float v = xr[c]; s += v * v; }
  #pragma unroll
  for (int off = 16; off > 0; off >>= 1) s += __shfl_xor(s, off, 32);
  __shared__ float red[8];
  if ((threadIdx.x & 31) == 0) red[threadIdx.x >> 5] = s;
  __syncthreads();
  float tot = 0.f;
  #pragma unroll
  for (int i = 0; i < 8; ++i) tot += red[i];
  const float inv = rsqrtf(tot * (1.0f / D_MODEL) + EPS);
  _Float16* outr = out + (size_t)row * D_MODEL;
  for (int c = threadIdx.x; c < D_MODEL; c += 256)
    outr[c] = (_Float16)(xr[c] * inv * g[c]);
}

// --------------------------- WMMA GEMM kernel ------------------------------
// C[M,N] = A[M,K] (f16 row-major) * B[N,K]^T (f16 row-major weights) +
//          optional f32 residual.  Block tile 128x64, 8 waves, wave tile 32x32.
// Tile staging via TDM tensor_load_to_lds (wave 0 issues the DMA descriptors,
// TENSORcnt tracks completion), double-buffered so DMA overlaps WMMA.
__global__ __launch_bounds__(256)
void gemm_f16_wmma_tdm_kernel(const _Float16* __restrict__ A,
                              const _Float16* __restrict__ Bm,
                              const float* __restrict__ resid,  // may be null
                              float* __restrict__ Cout,
                              int Mdim, int Ndim, int Kdim) {
  __shared__ _Float16 As[2][128 * 32];   // 2 x 8 KB
  __shared__ _Float16 Bs[2][64 * 32];    // 2 x 4 KB

  const int tid   = threadIdx.x;
  const int lane  = tid & 31;
  const int w     = tid >> 5;
  const int waveM = w & 3;            // 0..3  -> 32-row slice
  const int waveN = w >> 2;           // 0..1  -> 32-col slice
  const int blockM = blockIdx.y * 128;
  const int blockN = blockIdx.x * 64;
  const int lm = lane & 15;
  const int hf = lane >> 4;

  const unsigned int ldsA0 = (unsigned int)(uintptr_t)&As[0][0];
  const unsigned int ldsA1 = (unsigned int)(uintptr_t)&As[1][0];
  const unsigned int ldsB0 = (unsigned int)(uintptr_t)&Bs[0][0];
  const unsigned int ldsB1 = (unsigned int)(uintptr_t)&Bs[1][0];
  const unsigned long long gA =
      (unsigned long long)(uintptr_t)(A + (size_t)blockM * Kdim);
  const unsigned long long gB =
      (unsigned long long)(uintptr_t)(Bm + (size_t)blockN * Kdim);

  // Issue one 2-D TDM tile load: tile = tile_d1 rows x 32 halves, row stride
  // Kdim halves, data_size = 2 bytes.  D# per ISA 08_async_tensor §8.
  auto tdm_issue = [&](unsigned long long gbase, unsigned int lds_addr,
                       int tensor_d1, int tile_d1, int k0) {
    const unsigned long long ga = gbase + (unsigned long long)k0 * 2ull;
    v4u g0;
    g0.x = 1u;                                  // count=1, user descriptor
    g0.y = lds_addr;                            // LDS byte address
    g0.z = (unsigned int)ga;                    // global_addr[31:0]
    g0.w = (unsigned int)((ga >> 32) & 0x1FFFFFFu) | (2u << 30);  // type=2
    v8i g1;
    g1[0] = (int)(1u << 16);                    // wg_mask=0, data_size=1 (2B)
    g1[1] = (int)(((unsigned int)Kdim & 0xFFFFu) << 16);      // tensor_dim0 lo
    g1[2] = (int)(((unsigned int)Kdim >> 16) |
                  (((unsigned int)tensor_d1 & 0xFFFFu) << 16));
    g1[3] = (int)(((unsigned int)tensor_d1 >> 16) | (32u << 16)); // tile_dim0
    g1[4] = (int)(unsigned int)tile_d1;         // tile_dim1 (tile_dim2=0)
    g1[5] = (int)(unsigned int)Kdim;            // tensor_dim0_stride[31:0]
    g1[6] = 0;
    g1[7] = 0;
    const v4i z4 = (v4i){0, 0, 0, 0};
    const v8i z8 = (v8i){0, 0, 0, 0, 0, 0, 0, 0};
    __builtin_amdgcn_tensor_load_to_lds(g0, g1, z4, z4, z8, 0);
  };

  v8f c[2][2];
  #pragma unroll
  for (int i = 0; i < 2; ++i)
    #pragma unroll
    for (int j = 0; j < 2; ++j)
      c[i][j] = (v8f){0.f, 0.f, 0.f, 0.f, 0.f, 0.f, 0.f, 0.f};

  // Prologue: wave 0 kicks off the first pair of tile DMAs.
  if (w == 0) {
    tdm_issue(gA, ldsA0, Mdim, 128, 0);
    tdm_issue(gB, ldsB0, Ndim, 64, 0);
  }

  for (int it = 0, k0 = 0; k0 < Kdim; ++it, k0 += 32) {
    const int cur = it & 1;
    if (w == 0) __builtin_amdgcn_s_wait_tensorcnt(0);  // tiles[cur] landed
    __syncthreads();  // publish LDS to all waves; all done reading buf[cur^1]
    if (w == 0 && k0 + 32 < Kdim) {  // overlap next DMA with this compute
      tdm_issue(gA, cur ? ldsA0 : ldsA1, Mdim, 128, k0 + 32);
      tdm_issue(gB, cur ? ldsB0 : ldsB1, Ndim, 64, k0 + 32);
    }

    // Fragments per ISA 7.12.2 wave32 layouts.
    v16h a[2], b[2];
    #pragma unroll
    for (int s = 0; s < 2; ++s) {
      const int mrow = waveM * 32 + s * 16 + lm;
      const int kbA  = hf * 8;
      #pragma unroll
      for (int i = 0; i < 16; ++i)
        a[s][i] = As[cur][mrow * 32 + kbA + (i < 8 ? i : i + 8)];
      const int nrow = waveN * 32 + s * 16 + lm;
      const int kbB  = hf * 8;
      #pragma unroll
      for (int i = 0; i < 16; ++i)
        b[s][i] = Bs[cur][nrow * 32 + kbB + (i < 8 ? i : i + 8)];
    }
    #pragma unroll
    for (int i = 0; i < 2; ++i)
      #pragma unroll
      for (int j = 0; j < 2; ++j)
        c[i][j] = __builtin_amdgcn_wmma_f32_16x16x32_f16(
            false, a[i], false, b[j], (short)0, c[i][j], false, false);
  }

  // Epilogue: C/D layout -> row = r + 8*half, col = lane&15.
  #pragma unroll
  for (int i = 0; i < 2; ++i)
    #pragma unroll
    for (int j = 0; j < 2; ++j)
      #pragma unroll
      for (int r = 0; r < 8; ++r) {
        const int row = blockM + waveM * 32 + i * 16 + r + hf * 8;
        const int col = blockN + waveN * 32 + j * 16 + lm;
        float v = c[i][j][r];
        if (resid) v += resid[(size_t)row * Ndim + col];
        Cout[(size_t)row * Ndim + col] = v;
      }
}

// ---------------------- RoPE + head split (f32 -> f16) ---------------------
__global__ void rope_split_kernel(const float* __restrict__ qkv,
                                  _Float16* __restrict__ qh,
                                  _Float16* __restrict__ kh,
                                  _Float16* __restrict__ vh) {
  const int idx = blockIdx.x * 256 + threadIdx.x;  // B*H*T*32 = 2^21 total
  const int d2 = idx & 31;
  const int t  = (idx >> 5) & (TT - 1);
  const int h  = (idx >> 16) & (NUM_HEADS - 1);
  const int b  = idx >> 20;
  const size_t bt = (size_t)(b * TT + t);
  const float* base = qkv + bt * (3 * D_MODEL);
  const int ch = h * D_HEAD + 2 * d2;
  const float qe = base[ch],               qo = base[ch + 1];
  const float ke = base[D_MODEL + ch],     ko = base[D_MODEL + ch + 1];
  const float ve = base[2 * D_MODEL + ch], vo = base[2 * D_MODEL + ch + 1];
  const float inv = __expf(-logf(10000.f) * (float)(2 * d2) * (1.0f / D_HEAD));
  float sn, cs;
  __sincosf((float)t * inv, &sn, &cs);
  const size_t o = (((size_t)(b * NUM_HEADS + h) * TT + t) * D_HEAD) + 2 * d2;
  qh[o]     = (_Float16)(qe * cs - qo * sn);
  qh[o + 1] = (_Float16)(qe * sn + qo * cs);
  kh[o]     = (_Float16)(ke * cs - ko * sn);
  kh[o + 1] = (_Float16)(ke * sn + ko * cs);
  vh[o]     = (_Float16)ve;
  vh[o + 1] = (_Float16)vo;
}

// ----------------------- flash-style causal attention ----------------------
// grid = (B*H, T/128); block = 256 (8 waves); each wave owns 16 query rows.
__global__ __launch_bounds__(256)
void flash_attn_kernel(const _Float16* __restrict__ qh,
                       const _Float16* __restrict__ kh,
                       const _Float16* __restrict__ vh,
                       _Float16* __restrict__ attn /* [B,T,D_MODEL] */) {
  __shared__ _Float16 pbuf[8][16 * 32];   // per-wave P relayout buffer (8 KB)
  const int lane = threadIdx.x & 31;
  const int w    = threadIdx.x >> 5;
  const int bh   = blockIdx.x;            // 0..31
  const int b    = bh >> 4, h = bh & 15;
  const int qbase = blockIdx.y * 128 + w * 16;
  const int lm = lane & 15;
  const int hf = lane >> 4;

  const _Float16* Qb = qh + (size_t)bh * TT * D_HEAD;
  const _Float16* Kb = kh + (size_t)bh * TT * D_HEAD;
  const _Float16* Vb = vh + (size_t)bh * TT * D_HEAD;

  // Q as two A-fragments covering dh 0..31 / 32..63.
  v16h aq[2];
  {
    const _Float16* qrow = Qb + (size_t)(qbase + lm) * D_HEAD;
    #pragma unroll
    for (int f = 0; f < 2; ++f) {
      const int kb = f * 32 + hf * 8;
      #pragma unroll
      for (int i = 0; i < 16; ++i)
        aq[f][i] = qrow[kb + (i < 8 ? i : i + 8)];
    }
  }

  v8f O[4];
  #pragma unroll
  for (int d = 0; d < 4; ++d) O[d] = (v8f){0.f,0.f,0.f,0.f,0.f,0.f,0.f,0.f};
  float rowmax[8], rowsum[8];
  #pragma unroll
  for (int r = 0; r < 8; ++r) { rowmax[r] = -1e30f; rowsum[r] = 0.f; }

  const int kend = qbase + 16;            // keys needed (causal), exclusive
  for (int j = 0; j < kend; j += 32) {
    if (j + 32 < kend) {  // pull next K/V chunk toward the caches
      __builtin_prefetch(Kb + (size_t)(j + 32 + lane) * D_HEAD, 0, 1);
      __builtin_prefetch(Vb + (size_t)(j + 32 + lane) * D_HEAD, 0, 1);
    }
    // ---- two 16-key score tiles: S = (Q K^T) / sqrt(dh), causal-masked ----
    v8f s[2];
    #pragma unroll
    for (int kt = 0; kt < 2; ++kt) {
      const int kt0 = j + kt * 16;
      v16h bk[2];
      #pragma unroll
      for (int f = 0; f < 2; ++f) {
        int krow = kt0 + lm; krow = krow < TT ? krow : TT - 1;
        const _Float16* kp = Kb + (size_t)krow * D_HEAD + f * 32 + hf * 8;
        #pragma unroll
        for (int i = 0; i < 16; ++i) bk[f][i] = kp[i < 8 ? i : i + 8];
      }
      v8f z = (v8f){0.f,0.f,0.f,0.f,0.f,0.f,0.f,0.f};
      z = __builtin_amdgcn_wmma_f32_16x16x32_f16(false, aq[0], false, bk[0],
                                                 (short)0, z, false, false);
      z = __builtin_amdgcn_wmma_f32_16x16x32_f16(false, aq[1], false, bk[1],
                                                 (short)0, z, false, false);
      #pragma unroll
      for (int r = 0; r < 8; ++r) {
        const int qt = qbase + r + hf * 8;
        const int kk = kt0 + lm;
        s[kt][r] = (kk <= qt) ? z[r] * 0.125f : -1e30f;
      }
    }
    // ---- online softmax (rows live in 16-lane groups) ---------------------
    #pragma unroll
    for (int r = 0; r < 8; ++r) {
      float m2 = fmaxf(s[0][r], s[1][r]);
      m2 = fmaxf(m2, __shfl_xor(m2, 1, 16));
      m2 = fmaxf(m2, __shfl_xor(m2, 2, 16));
      m2 = fmaxf(m2, __shfl_xor(m2, 4, 16));
      m2 = fmaxf(m2, __shfl_xor(m2, 8, 16));
      const float nm   = fmaxf(rowmax[r], m2);
      const float corr = __expf(rowmax[r] - nm);
      rowmax[r] = nm;
      const float p0 = __expf(s[0][r] - nm);
      const float p1 = __expf(s[1][r] - nm);
      s[0][r] = p0; s[1][r] = p1;
      float ps = p0 + p1;
      ps += __shfl_xor(ps, 1, 16);
      ps += __shfl_xor(ps, 2, 16);
      ps += __shfl_xor(ps, 4, 16);
      ps += __shfl_xor(ps, 8, 16);
      rowsum[r] = rowsum[r] * corr + ps;
      #pragma unroll
      for (int d = 0; d < 4; ++d) O[d][r] *= corr;
    }
    // ---- re-layout P (C-layout) -> A-fragment via per-wave LDS ------------
    #pragma unroll
    for (int kt = 0; kt < 2; ++kt)
      #pragma unroll
      for (int r = 0; r < 8; ++r)
        pbuf[w][(r + hf * 8) * 32 + kt * 16 + lm] = (_Float16)s[kt][r];
    v16h pf;
    {
      const int kb = hf * 8;
      #pragma unroll
      for (int i = 0; i < 16; ++i)
        pf[i] = pbuf[w][lm * 32 + kb + (i < 8 ? i : i + 8)];
    }
    // ---- O += P * V (4 x 16-col dh slices) --------------------------------
    #pragma unroll
    for (int d = 0; d < 4; ++d) {
      v16h bv;
      #pragma unroll
      for (int i = 0; i < 16; ++i) {
        int vr = j + hf * 8 + (i < 8 ? i : i + 8); vr = vr < TT ? vr : TT - 1;
        bv[i] = Vb[(size_t)vr * D_HEAD + d * 16 + lm];
      }
      O[d] = __builtin_amdgcn_wmma_f32_16x16x32_f16(false, pf, false, bv,
                                                    (short)0, O[d], false, false);
    }
  }

  // Normalize and scatter to [B,T,C] f16 (heads interleaved back into C).
  #pragma unroll
  for (int d = 0; d < 4; ++d)
    #pragma unroll
    for (int r = 0; r < 8; ++r) {
      const int t   = qbase + r + hf * 8;
      const int chn = h * D_HEAD + d * 16 + lm;
      attn[(size_t)(b * TT + t) * D_MODEL + chn] =
          (_Float16)(O[d][r] / rowsum[r]);
    }
}

// ------------------------------- SwiGLU ------------------------------------
__global__ void swiglu_kernel(const float* __restrict__ a,
                              const float* __restrict__ bg,
                              _Float16* __restrict__ y, int n) {
  int i = blockIdx.x * 256 + threadIdx.x;
  if (i < n) {
    const float av = a[i];
    const float sv = av / (1.f + __expf(-av));
    y[i] = (_Float16)(sv * bg[i]);
  }
}

// ------------------------------ launcher -----------------------------------

extern "C" void kernel_launch(void* const* d_in, const int* in_sizes, int n_in,
                              void* d_out, int out_size, void* d_ws, size_t ws_size,
                              hipStream_t stream) {
  (void)in_sizes; (void)n_in; (void)out_size; (void)ws_size;
  const float* x     = (const float*)d_in[0];
  const float* W_qkv = (const float*)d_in[1];
  const float* W_o   = (const float*)d_in[2];
  const float* g1    = (const float*)d_in[3];
  const float* g2    = (const float*)d_in[4];
  const float* W1    = (const float*)d_in[5];
  const float* W2    = (const float*)d_in[6];
  const float* W3    = (const float*)d_in[7];
  float* out = (float*)d_out;

  // workspace bump allocator (256B aligned)
  char* p = (char*)d_ws;
  auto alloc = [&](size_t bytes) -> void* {
    void* r = (void*)p;
    p += (bytes + 255) & ~(size_t)255;
    return r;
  };
  _Float16* h16    = (_Float16*)alloc((size_t)M_TOK * D_MODEL * 2);
  _Float16* wqkv16 = (_Float16*)alloc((size_t)3 * D_MODEL * D_MODEL * 2);
  _Float16* wo16   = (_Float16*)alloc((size_t)D_MODEL * D_MODEL * 2);
  _Float16* w1_16  = (_Float16*)alloc((size_t)D_FF * D_MODEL * 2);
  _Float16* w2_16  = (_Float16*)alloc((size_t)D_MODEL * D_FF * 2);
  _Float16* w3_16  = (_Float16*)alloc((size_t)D_FF * D_MODEL * 2);
  float*    qkv    = (float*)   alloc((size_t)M_TOK * 3 * D_MODEL * 4);
  _Float16* qh     = (_Float16*)alloc((size_t)M_TOK * D_MODEL * 2);
  _Float16* kh     = (_Float16*)alloc((size_t)M_TOK * D_MODEL * 2);
  _Float16* vh     = (_Float16*)alloc((size_t)M_TOK * D_MODEL * 2);
  _Float16* attn16 = (_Float16*)alloc((size_t)M_TOK * D_MODEL * 2);
  float*    a_f32  = (float*)   alloc((size_t)M_TOK * D_FF * 4);
  float*    b_f32  = (float*)   alloc((size_t)M_TOK * D_FF * 4);
  _Float16* y16    = (_Float16*)alloc((size_t)M_TOK * D_FF * 2);

  const int nt = 256;
  auto cvt = [&](const float* src, _Float16* dst, int n) {
    f32_to_f16_kernel<<<(n + nt - 1) / nt, nt, 0, stream>>>(src, dst, n);
  };
  // 1) weight precision conversion
  cvt(W_qkv, wqkv16, 3 * D_MODEL * D_MODEL);
  cvt(W_o,   wo16,   D_MODEL * D_MODEL);
  cvt(W1,    w1_16,  D_FF * D_MODEL);
  cvt(W2,    w2_16,  D_MODEL * D_FF);
  cvt(W3,    w3_16,  D_FF * D_MODEL);

  // 2) h = rmsnorm(x, g1)
  rmsnorm_f16_kernel<<<M_TOK, nt, 0, stream>>>(x, g1, h16);

  // 3) qkv = h @ W_qkv^T   [4096 x 3072]
  gemm_f16_wmma_tdm_kernel<<<dim3(3 * D_MODEL / 64, M_TOK / 128), nt, 0, stream>>>(
      h16, wqkv16, nullptr, qkv, M_TOK, 3 * D_MODEL, D_MODEL);

  // 4) RoPE + head split
  rope_split_kernel<<<(BB * NUM_HEADS * TT * 32) / nt, nt, 0, stream>>>(
      qkv, qh, kh, vh);

  // 5) causal flash attention
  flash_attn_kernel<<<dim3(BB * NUM_HEADS, TT / 128), nt, 0, stream>>>(
      qh, kh, vh, attn16);

  // 6) x2 = x + attn @ W_o^T   (written to d_out)
  gemm_f16_wmma_tdm_kernel<<<dim3(D_MODEL / 64, M_TOK / 128), nt, 0, stream>>>(
      attn16, wo16, x, out, M_TOK, D_MODEL, D_MODEL);

  // 7) h2 = rmsnorm(x2, g2)   (reuse h16)
  rmsnorm_f16_kernel<<<M_TOK, nt, 0, stream>>>(out, g2, h16);

  // 8) a = h2 @ W1^T ; b = h2 @ W3^T   [4096 x 4096]
  gemm_f16_wmma_tdm_kernel<<<dim3(D_FF / 64, M_TOK / 128), nt, 0, stream>>>(
      h16, w1_16, nullptr, a_f32, M_TOK, D_FF, D_MODEL);
  gemm_f16_wmma_tdm_kernel<<<dim3(D_FF / 64, M_TOK / 128), nt, 0, stream>>>(
      h16, w3_16, nullptr, b_f32, M_TOK, D_FF, D_MODEL);

  // 9) y = silu(a) * b
  swiglu_kernel<<<(M_TOK * D_FF) / nt, nt, 0, stream>>>(
      a_f32, b_f32, y16, M_TOK * D_FF);

  // 10) out = x2 + y @ W2^T
  gemm_f16_wmma_tdm_kernel<<<dim3(D_MODEL / 64, M_TOK / 128), nt, 0, stream>>>(
      y16, w2_16, out, out, M_TOK, D_MODEL, D_FF);
}
